// MultiHeadLatentAttention_70738111365895
// MI455X (gfx1250) — matrix operand order repaired and checked by reading, not compiled, not run on hardware
//
#include <hip/hip_runtime.h>
#include <hip/hip_bf16.h>

typedef __bf16 bf16;
typedef __attribute__((ext_vector_type(16))) __bf16 v16bf;
typedef __attribute__((ext_vector_type(8)))  __bf16 v8bf;
typedef __attribute__((ext_vector_type(8)))  float  v8f;
typedef __attribute__((ext_vector_type(4)))  unsigned int v4u;
typedef __attribute__((ext_vector_type(8)))  int v8i;
typedef __attribute__((ext_vector_type(4)))  int v4i;

#define HIDDEN_D 2048
#define SEQ_LEN  2048
#define N_HEADS  32
#define HEAD_D   64
#define ROPE_D   32
#define LATENT_D 512

__device__ __forceinline__ v16bf cat16(v8bf lo, v8bf hi) {
    return __builtin_shufflevector(lo, hi, 0,1,2,3,4,5,6,7,8,9,10,11,12,13,14,15);
}
__device__ __forceinline__ v8f zero8() {
    v8f z;
#pragma unroll
    for (int r = 0; r < 8; ++r) z[r] = 0.0f;
    return z;
}
__device__ __forceinline__ v8f wmma_bf16(v16bf a, v16bf b, v8f c) {
    return __builtin_amdgcn_wmma_f32_16x16x32_bf16(false, a, false, b, (short)0, c, false, false);
}

// ------------------------------------------------------------------ TDM tile load
// DMA a 2D bf16 tile (tile_dim0=32 contiguous elems x tile_dim1=128 rows) from a
// row-major tensor (row length = stride = K elems, M rows for OOB) into LDS,
// padding each 16-DWORD row with 4 DWORDs so LDS rows land on a 40-element pitch.
// D# bitfields per CDNA5 ISA 8.3/8.4. Issued by one wave; tracked by TENSORcnt.
__device__ __forceinline__ void tdm_load_tile_a(const bf16* gptr, unsigned lds_byte_addr,
                                                int K, int M)
{
    unsigned long long ga = (unsigned long long)(size_t)gptr;
    // group0: count=1 | lds_addr | global_addr[31:0] | global_addr[56:32] + type=2
    v4u g0 = { 1u, lds_byte_addr, (unsigned)ga, (unsigned)(ga >> 32) | (2u << 30) };
    // group1 word0: data_size=1 (2B) | pad_enable | pad_interval=3 (16 DW) | pad_amount=3 (4 DW)
    v8i g1 = { (int)0x06D10000,
               (int)(((unsigned)K & 0xFFFFu) << 16),                              // tensor_dim0 lo16<<16
               (int)((((unsigned)K >> 16) & 0xFFFFu) | (((unsigned)M & 0xFFFFu) << 16)), // dim0 hi | dim1 lo
               (int)((((unsigned)M >> 16) & 0xFFFFu) | (32u << 16)),              // dim1 hi | tile_dim0=32
               128,                                                               // tile_dim1=128, tile_dim2=0
               K,                                                                 // tensor_dim0_stride lo32
               0, 0 };
    v4i g2 = {0, 0, 0, 0};
    v4i g3 = {0, 0, 0, 0};
#if defined(__clang_major__) && (__clang_major__ >= 23)
    v8i gz = {0, 0, 0, 0, 0, 0, 0, 0};
    __builtin_amdgcn_tensor_load_to_lds(g0, g1, g2, g3, gz, 0);
#else
    __builtin_amdgcn_tensor_load_to_lds(g0, g1, g2, g3, 0);
#endif
}

// ---------------------------------------------------------------- fp32 -> bf16
__global__ void cvt_f32_bf16(const float* __restrict__ in, bf16* __restrict__ out, int n) {
    int i = blockIdx.x * blockDim.x + threadIdx.x;
    if (i < n) out[i] = (bf16)in[i];
}

// ---------------------------------------------------------------- tiled WMMA GEMM
// C[M,N] = A[M,K](bf16,row-major) @ B[K,N](bf16,row-major)
// Block tile 128x128x32, 4 waves. A tile staged by the Tensor Data Mover
// (async DMA + LDS padding), B tile staged transposed by the waves.
template <bool OUT_BF16>
__global__ __launch_bounds__(128)
void gemm_bf16_wmma(const bf16* __restrict__ A, const bf16* __restrict__ B,
                    void* __restrict__ Cout, int M, int N, int K)
{
    __shared__ bf16 As[128][40];   // [m][k], 40-elem pitch (TDM pads 32->40)
    __shared__ bf16 BsT[128][40];  // [n][k], B transposed, padded

    const int tid  = threadIdx.x;
    const int lane = tid & 31;
    const int w    = tid >> 5;
    const int l16  = lane & 15;
    const int hi   = lane >> 4;
    const int wm   = (w >> 1) * 64;
    const int wn   = (w & 1) * 64;
    const int bm   = blockIdx.y * 128;
    const int bn   = blockIdx.x * 128;

    v8f acc[4][4];
#pragma unroll
    for (int i = 0; i < 4; ++i)
#pragma unroll
        for (int j = 0; j < 4; ++j) acc[i][j] = zero8();

    for (int k0 = 0; k0 < K; k0 += 32) {
        // stage A tile via TDM (wave 0 issues; EXEC-independent)
        if (w == 0)
            tdm_load_tile_a(A + (size_t)bm * K + k0, (unsigned)(size_t)&As[0][0], K, M);

        // stage B tile transposed: 32x128 read coalesced, scattered to BsT[n][k]
#pragma unroll
        for (int c = 0; c < 4; ++c) {
            int chunk = c * 128 + tid;
            int kr = chunk >> 4, nc = (chunk & 15) * 8;
            v8bf vv = *(const v8bf*)&B[(size_t)(k0 + kr) * N + bn + nc];
#pragma unroll
            for (int e = 0; e < 8; ++e) BsT[nc + e][kr] = vv[e];
            if (k0 + 32 < K)
                __builtin_prefetch(&B[(size_t)(k0 + 32 + kr) * N + bn + nc], 0, 1);
        }
        __builtin_amdgcn_s_wait_tensorcnt(0);
        __syncthreads();

        v16bf af[4];
#pragma unroll
        for (int i = 0; i < 4; ++i) {
            const bf16* p = &As[wm + i * 16 + l16][0];
            af[i] = cat16(*(const v8bf*)(p + 8 * hi),
                          *(const v8bf*)(p + 16 + 8 * hi));
        }
#pragma unroll
        for (int j = 0; j < 4; ++j) {
            const bf16* p = &BsT[wn + j * 16 + l16][0];
            v16bf bfr = cat16(*(const v8bf*)(p + 8 * hi),
                              *(const v8bf*)(p + 16 + 8 * hi));
#pragma unroll
            for (int i = 0; i < 4; ++i) acc[i][j] = wmma_bf16(af[i], bfr, acc[i][j]);
        }
        __syncthreads();
    }

#pragma unroll
    for (int i = 0; i < 4; ++i)
#pragma unroll
        for (int j = 0; j < 4; ++j)
#pragma unroll
            for (int r = 0; r < 8; ++r) {
                int row = bm + wm + i * 16 + r + 8 * hi;
                int col = bn + wn + j * 16 + l16;
                float v = acc[i][j][r];
                if (OUT_BF16) ((bf16*)Cout)[(size_t)row * N + col] = (bf16)v;
                else          ((float*)Cout)[(size_t)row * N + col] = v;
            }
}

// ---------------------------------------------------------------- RoPE + concat
// Builds K = [k_proj | rope(k_rope)], Q = 0.125 * [q_proj | rope(q_rope)]
__global__ void rope_concat(const float* __restrict__ kp, const float* __restrict__ qp,
                            const float* __restrict__ kr, const float* __restrict__ qr,
                            bf16* __restrict__ Kc, bf16* __restrict__ Qc)
{
    int t = blockIdx.x * blockDim.x + threadIdx.x;
    if (t >= SEQ_LEN * N_HEADS * 16) return;
    int i = t & 15;
    int h = (t >> 4) & (N_HEADS - 1);
    int s = t >> 9;
    size_t ib = (size_t)s * (N_HEADS * ROPE_D) + h * ROPE_D;
    size_t ob = (size_t)s * HIDDEN_D + h * HEAD_D;
    float ang = (float)s * __powf(10000.0f, -(float)i * (1.0f / 16.0f));
    float c, sn;
    __sincosf(ang, &sn, &c);
    {
        float x1 = kr[ib + i], x2 = kr[ib + i + 16];
        Kc[ob + i]           = (bf16)kp[ib + i];
        Kc[ob + i + 16]      = (bf16)kp[ib + i + 16];
        Kc[ob + 32 + i]      = (bf16)(x1 * c - x2 * sn);
        Kc[ob + 32 + i + 16] = (bf16)(x2 * c + x1 * sn);
    }
    {
        const float sc = 0.125f; // 1/sqrt(64) folded into Q
        float x1 = qr[ib + i], x2 = qr[ib + i + 16];
        Qc[ob + i]           = (bf16)(qp[ib + i] * sc);
        Qc[ob + i + 16]      = (bf16)(qp[ib + i + 16] * sc);
        Qc[ob + 32 + i]      = (bf16)((x1 * c - x2 * sn) * sc);
        Qc[ob + 32 + i + 16] = (bf16)((x2 * c + x1 * sn) * sc);
    }
}

// ---------------------------------------------------------------- fused flash attention
// grid (SEQ/128, HEADS), 128 threads (4 waves). Each wave: 32 query rows x 64 dims.
// Key blocks of 32 with online softmax. QK^T and P@V via WMMA bf16.
__global__ __launch_bounds__(128)
void mla_attention(const bf16* __restrict__ Qc, const bf16* __restrict__ Kc,
                   const bf16* __restrict__ Vc, bf16* __restrict__ Ob)
{
    __shared__ bf16 VT[64][40];       // V tile transposed: [dim][key]
    __shared__ bf16 Pt[4][32][40];    // per-wave P tile round-trip

    const int tid  = threadIdx.x;
    const int lane = tid & 31;
    const int w    = tid >> 5;
    const int l16  = lane & 15;
    const int hi   = lane >> 4;
    const int qb   = blockIdx.x;
    const int h    = blockIdx.y;
    const int qrow = qb * 128 + w * 32;
    const float NEG = -3.0e38f;

    // persistent Q fragments (A-layout, loaded straight from global)
    v16bf qf[2][2];
#pragma unroll
    for (int i = 0; i < 2; ++i)
#pragma unroll
        for (int ks = 0; ks < 2; ++ks) {
            const bf16* p = Qc + (size_t)(qrow + i * 16 + l16) * HIDDEN_D + h * HEAD_D + ks * 32;
            qf[i][ks] = cat16(*(const v8bf*)(p + 16 * hi),
                              *(const v8bf*)(p + 16 * hi + 8));
        }

    v8f o[2][4];
    float mrun[2][8], lrun[2][8];
#pragma unroll
    for (int i = 0; i < 2; ++i) {
#pragma unroll
        for (int t = 0; t < 4; ++t) o[i][t] = zero8();
#pragma unroll
        for (int r = 0; r < 8; ++r) { mrun[i][r] = NEG; lrun[i][r] = 0.0f; }
    }

    const int nkb = (qb + 1) * 4; // causal: cover up to workgroup's last row
    for (int kb = 0; kb < nkb; ++kb) {
        const int kbase = kb * 32;
        // cooperative staging of V tile (32 keys x 64 dims), transposed into LDS
#pragma unroll
        for (int c = 0; c < 2; ++c) {
            int chunk = c * 128 + tid;
            int kr = chunk >> 3, dc = (chunk & 7) * 8;
            v8bf vv = *(const v8bf*)&Vc[(size_t)(kbase + kr) * HIDDEN_D + h * HEAD_D + dc];
#pragma unroll
            for (int e = 0; e < 8; ++e) VT[dc + e][kr] = vv[e];
        }
        // prefetch next key block (K and V streams)
        if (kb + 1 < nkb) {
            const size_t nb = (size_t)(kbase + 32 + lane) * HIDDEN_D + h * HEAD_D + (tid >> 5) * 16;
            __builtin_prefetch(&Vc[nb], 0, 1);
            __builtin_prefetch(&Kc[nb], 0, 1);
        }
        __syncthreads();

        // S = Q K^T  (K rows act as B columns -> contiguous per-lane loads)
        v8f s[2][2];
#pragma unroll
        for (int j = 0; j < 2; ++j) {
            const bf16* p = Kc + (size_t)(kbase + j * 16 + l16) * HIDDEN_D + h * HEAD_D + 16 * hi;
            v16bf kf0 = cat16(*(const v8bf*)p,        *(const v8bf*)(p + 8));
            v16bf kf1 = cat16(*(const v8bf*)(p + 32), *(const v8bf*)(p + 40));
#pragma unroll
            for (int i = 0; i < 2; ++i) {
                v8f z = wmma_bf16(qf[i][0], kf0, zero8());
                s[i][j] = wmma_bf16(qf[i][1], kf1, z);
            }
        }

        // causal mask + online softmax (rows live in 16-lane groups)
#pragma unroll
        for (int i = 0; i < 2; ++i) {
            float rmax[8];
#pragma unroll
            for (int r = 0; r < 8; ++r) {
                int row = qrow + i * 16 + r + 8 * hi;
                float a = NEG;
#pragma unroll
                for (int j = 0; j < 2; ++j) {
                    int col = kbase + j * 16 + l16;
                    float v = s[i][j][r];
                    if (col > row) v = NEG;
                    s[i][j][r] = v;
                    a = fmaxf(a, v);
                }
                a = fmaxf(a, __shfl_xor(a, 1, 32));
                a = fmaxf(a, __shfl_xor(a, 2, 32));
                a = fmaxf(a, __shfl_xor(a, 4, 32));
                a = fmaxf(a, __shfl_xor(a, 8, 32));
                rmax[r] = a;
            }
#pragma unroll
            for (int r = 0; r < 8; ++r) {
                float mnew  = fmaxf(mrun[i][r], rmax[r]);
                float scale = __expf(mrun[i][r] - mnew);
                mrun[i][r]  = mnew;
                float rsum  = 0.0f;
#pragma unroll
                for (int j = 0; j < 2; ++j) {
                    float pv = __expf(s[i][j][r] - mnew);
                    s[i][j][r] = pv;
                    rsum += pv;
                }
                rsum += __shfl_xor(rsum, 1, 32);
                rsum += __shfl_xor(rsum, 2, 32);
                rsum += __shfl_xor(rsum, 4, 32);
                rsum += __shfl_xor(rsum, 8, 32);
                lrun[i][r] = lrun[i][r] * scale + rsum;
#pragma unroll
                for (int t = 0; t < 4; ++t) o[i][t][r] *= scale;
            }
            // spill P (C-layout) to LDS row-major for A-layout reload
#pragma unroll
            for (int j = 0; j < 2; ++j)
#pragma unroll
                for (int r = 0; r < 8; ++r)
                    Pt[w][i * 16 + r + 8 * hi][j * 16 + l16] = (bf16)s[i][j][r];
        }

        // O += P @ V
#pragma unroll
        for (int i = 0; i < 2; ++i) {
            const bf16* pp = &Pt[w][i * 16 + l16][0];
            v16bf pf = cat16(*(const v8bf*)(pp + 16 * hi),
                             *(const v8bf*)(pp + 16 * hi + 8));
#pragma unroll
            for (int t = 0; t < 4; ++t) {
                const bf16* vp = &VT[t * 16 + l16][16 * hi];
                v16bf vf = cat16(*(const v8bf*)vp, *(const v8bf*)(vp + 8));
                o[i][t] = wmma_bf16(pf, vf, o[i][t]);
            }
        }
        __syncthreads();
    }

    // normalize and store (bf16, feeds the W_o GEMM)
#pragma unroll
    for (int i = 0; i < 2; ++i)
#pragma unroll
        for (int t = 0; t < 4; ++t)
#pragma unroll
            for (int r = 0; r < 8; ++r) {
                int row = qrow + i * 16 + r + 8 * hi;
                int col = h * HEAD_D + t * 16 + l16;
                Ob[(size_t)row * HIDDEN_D + col] = (bf16)(o[i][t][r] / lrun[i][r]);
            }
}

// ---------------------------------------------------------------- host
extern "C" void kernel_launch(void* const* d_in, const int* in_sizes, int n_in,
                              void* d_out, int out_size, void* d_ws, size_t ws_size,
                              hipStream_t stream)
{
    (void)in_sizes; (void)n_in; (void)out_size; (void)ws_size;
    const float* x    = (const float*)d_in[0];
    const float* Wkvd = (const float*)d_in[1];
    const float* Wqd  = (const float*)d_in[2];
    const float* Wku  = (const float*)d_in[3];
    const float* Wvu  = (const float*)d_in[4];
    const float* Wqu  = (const float*)d_in[5];
    const float* Wrk  = (const float*)d_in[6];
    const float* Wrq  = (const float*)d_in[7];
    const float* Wo   = (const float*)d_in[8];

    char* base = (char*)d_ws;
    size_t off = 0;
    auto alloc = [&](size_t elems, size_t esz) -> void* {
        void* p = base + off;
        off += (elems * esz + 255) & ~(size_t)255;
        return p;
    };

    bf16* x_bf    = (bf16*)alloc((size_t)SEQ_LEN * HIDDEN_D, 2);
    bf16* Wkvd_bf = (bf16*)alloc((size_t)HIDDEN_D * LATENT_D, 2);
    bf16* Wqd_bf  = (bf16*)alloc((size_t)HIDDEN_D * LATENT_D, 2);
    bf16* Wku_bf  = (bf16*)alloc((size_t)LATENT_D * 1024, 2);
    bf16* Wvu_bf  = (bf16*)alloc((size_t)LATENT_D * HIDDEN_D, 2);
    bf16* Wqu_bf  = (bf16*)alloc((size_t)LATENT_D * 1024, 2);
    bf16* Wrk_bf  = (bf16*)alloc((size_t)HIDDEN_D * 1024, 2);
    bf16* Wrq_bf  = (bf16*)alloc((size_t)LATENT_D * 1024, 2);
    bf16* Wo_bf   = (bf16*)alloc((size_t)HIDDEN_D * HIDDEN_D, 2);

    bf16*  kvd_bf = (bf16*)alloc((size_t)SEQ_LEN * LATENT_D, 2);
    bf16*  qd_bf  = (bf16*)alloc((size_t)SEQ_LEN * LATENT_D, 2);
    float* kproj  = (float*)alloc((size_t)SEQ_LEN * 1024, 4);
    float* qproj  = (float*)alloc((size_t)SEQ_LEN * 1024, 4);
    float* krope  = (float*)alloc((size_t)SEQ_LEN * 1024, 4);
    float* qrope  = (float*)alloc((size_t)SEQ_LEN * 1024, 4);
    bf16*  v_bf   = (bf16*)alloc((size_t)SEQ_LEN * HIDDEN_D, 2);
    bf16*  K_bf   = (bf16*)alloc((size_t)SEQ_LEN * HIDDEN_D, 2);
    bf16*  Q_bf   = (bf16*)alloc((size_t)SEQ_LEN * HIDDEN_D, 2);
    bf16*  att_bf = (bf16*)alloc((size_t)SEQ_LEN * HIDDEN_D, 2);

    auto cvt = [&](const float* in, bf16* out, int n) {
        cvt_f32_bf16<<<(n + 255) / 256, 256, 0, stream>>>(in, out, n);
    };
    cvt(x,    x_bf,    SEQ_LEN * HIDDEN_D);
    cvt(Wkvd, Wkvd_bf, HIDDEN_D * LATENT_D);
    cvt(Wqd,  Wqd_bf,  HIDDEN_D * LATENT_D);
    cvt(Wku,  Wku_bf,  LATENT_D * 1024);
    cvt(Wvu,  Wvu_bf,  LATENT_D * HIDDEN_D);
    cvt(Wqu,  Wqu_bf,  LATENT_D * 1024);
    cvt(Wrk,  Wrk_bf,  HIDDEN_D * 1024);
    cvt(Wrq,  Wrq_bf,  LATENT_D * 1024);
    cvt(Wo,   Wo_bf,   HIDDEN_D * HIDDEN_D);

    auto gemm_b = [&](const bf16* A, const bf16* B, bf16* C, int M, int N, int K) {
        gemm_bf16_wmma<true><<<dim3(N / 128, M / 128), 128, 0, stream>>>(A, B, (void*)C, M, N, K);
    };
    auto gemm_f = [&](const bf16* A, const bf16* B, float* C, int M, int N, int K) {
        gemm_bf16_wmma<false><<<dim3(N / 128, M / 128), 128, 0, stream>>>(A, B, (void*)C, M, N, K);
    };

    gemm_b(x_bf,   Wkvd_bf, kvd_bf, SEQ_LEN, LATENT_D, HIDDEN_D);   // kv_d
    gemm_b(x_bf,   Wqd_bf,  qd_bf,  SEQ_LEN, LATENT_D, HIDDEN_D);   // q_d
    gemm_f(kvd_bf, Wku_bf,  kproj,  SEQ_LEN, 1024,     LATENT_D);   // k_proj
    gemm_f(qd_bf,  Wqu_bf,  qproj,  SEQ_LEN, 1024,     LATENT_D);   // q_proj
    gemm_b(kvd_bf, Wvu_bf,  v_bf,   SEQ_LEN, HIDDEN_D, LATENT_D);   // v
    gemm_f(x_bf,   Wrk_bf,  krope,  SEQ_LEN, 1024,     HIDDEN_D);   // k_rope
    gemm_f(qd_bf,  Wrq_bf,  qrope,  SEQ_LEN, 1024,     LATENT_D);   // q_rope

    {
        int n = SEQ_LEN * N_HEADS * 16;
        rope_concat<<<(n + 255) / 256, 256, 0, stream>>>(kproj, qproj, krope, qrope, K_bf, Q_bf);
    }

    mla_attention<<<dim3(SEQ_LEN / 128, N_HEADS), 128, 0, stream>>>(Q_bf, K_bf, v_bf, att_bf);

    gemm_f(att_bf, Wo_bf, (float*)d_out, SEQ_LEN, HIDDEN_D, HIDDEN_D); // @ W_o
}
